// EncoderLayer_37701222924438
// MI455X (gfx1250) — matrix operand order repaired and checked by reading, not compiled, not run on hardware
//
#include <hip/hip_runtime.h>
#include <hip/hip_bf16.h>

// ---------------------------------------------------------------------------
// Problem constants (fixed by the reference)
// ---------------------------------------------------------------------------
constexpr int kB   = 4;
constexpr int kS   = 1024;
constexpr int kD   = 1024;
constexpr int kH   = 16;
constexpr int kDK  = 64;     // kD / kH
constexpr int kDFF = 4096;
#define LN_EPS 1e-5f

typedef __attribute__((ext_vector_type(16))) _Float16 v16h;
typedef __attribute__((ext_vector_type(8)))  _Float16 v8h;
typedef __attribute__((ext_vector_type(8)))  float    v8f;
typedef __attribute__((ext_vector_type(16))) float    v16f;

// Async global->LDS staging (CDNA5 GLOBAL_LOAD_ASYNC_TO_LDS_B128, ASYNCcnt),
// guarded so we fall back to VGPR staging if the toolchain lacks the builtin.
#if defined(__HIP_DEVICE_COMPILE__) && \
    __has_builtin(__builtin_amdgcn_global_load_async_to_lds_b128)
#define USE_ASYNC_LDS 1
#else
#define USE_ASYNC_LDS 0
#endif

#if USE_ASYNC_LDS
// Builtin prototype (from the compiler diagnostic):
//   void __builtin_amdgcn_global_load_async_to_lds_b128(
//       int4 addrspace(1)* gsrc, int4 addrspace(3)* ldst,
//       imm int offset, imm int cpol)
typedef int v4i_b128 __attribute__((vector_size(16)));
typedef __attribute__((address_space(1))) v4i_b128* gas_v4i;
typedef __attribute__((address_space(3))) v4i_b128* las_v4i;

template <int OFF>
__device__ __forceinline__ void async_copy_b128(const void* g, void* l) {
    __builtin_amdgcn_global_load_async_to_lds_b128(
        (gas_v4i)(g), (las_v4i)(l), OFF, 0);
}
__device__ __forceinline__ void async_wait_all() {
#if __has_builtin(__builtin_amdgcn_s_wait_asynccnt)
    __builtin_amdgcn_s_wait_asynccnt(0);
#else
    asm volatile("s_wait_asynccnt 0" ::: "memory");
#endif
}
#else
__device__ __forceinline__ void async_wait_all() {}
#endif

// ---------------------------------------------------------------------------
// Generic tiled WMMA GEMM:  C = alpha * (A[M,K] @ Bt[N,K]^T) + bias
//   - A is f16 (AF32=false) or f32 converted to f16 during LDS staging (AF32=true)
//   - Bt is the f16 transposed B operand, row-major [N,K]
//   - optional fp32 output C32 (ldc32), optional f16 output Ch with layout modes:
//       chmode 0: row-major [m, n] with leading dim ldch
//       chmode 1: head-split  [b, h, s, dk]   (m = b*S+s, n = h*DK+dk)
//       chmode 2: head-split-T [b, h, dk, s]  (for V^T)
//   - batched over blockIdx.z with element strides; Ch batch offset uses
//     (z/zdiv)*strideCh1 + (z%zdiv)*strideCh2 (for head-merged ctx output)
// Tiling: BM=128, BK=32, BN template (128 or 64); 256 threads = 8 wave32 waves
// arranged 4 (M) x 2 (N); each wave owns 32 x (BN/2) of C => 2 x (BN/32)
// v_wmma_f32_16x16x32_f16 accumulator tiles.
// ---------------------------------------------------------------------------
template <int BN, bool AF32>
__global__ __launch_bounds__(256) void gemm_f16_wmma(
    const void* __restrict__ Ain, const _Float16* __restrict__ Bt,
    const float* __restrict__ bias,
    float* __restrict__ C32, int ldc32, long long strideCz,
    _Float16* __restrict__ Ch, int ldch,
    long long strideCh1, long long strideCh2, int zdiv, int chmode,
    int K, long long strideAz, long long strideBz,
    float alpha, int relu)
{
    constexpr int BM  = 128;
    constexpr int BK  = 32;
    constexpr int LDT = BK + 8;   // padded row (halfs); 80B keeps 16B alignment

    __shared__ _Float16 As[2][BM * LDT];
    __shared__ _Float16 Bs[2][BN * LDT];

    const int tid = threadIdx.x;
    const int z   = blockIdx.z;

    const long long zA = (long long)z * strideAz;
    const long long zB = (long long)z * strideBz;
    if (C32) C32 += (long long)z * strideCz;
    if (Ch)  Ch  += (long long)(z / zdiv) * strideCh1 +
                    (long long)(z % zdiv) * strideCh2;

    const int m0 = blockIdx.y * BM;
    const int n0 = blockIdx.x * BN;

    // --- staging coordinates -------------------------------------------------
    const int ar = tid >> 1;          // A: 128 rows x 32 cols, 16 halfs/thread
    const int ac = (tid & 1) << 4;
    int br, bc;
    if constexpr (BN == 128) { br = tid >> 1; bc = (tid & 1) << 4; }
    else                     { br = tid >> 2; bc = (tid & 3) << 3; }

    const _Float16* Ah = (const _Float16*)Ain + zA + (long long)(m0 + ar) * K;
    const float*    Af = (const float*)Ain    + zA + (long long)(m0 + ar) * K;
    const _Float16* Bp = Bt + zB + (long long)(n0 + br) * K;

    auto stage = [&](int kt, int buf) {
        const int k0 = kt * BK;
        // ---- A tile ----
        if constexpr (AF32) {
            // f32 source: convert in registers, then LDS store
            v16f a32 = *(const v16f*)(Af + k0 + ac);
            v16h av;
            #pragma unroll
            for (int i = 0; i < 16; ++i) av[i] = (_Float16)a32[i];
            *(v16h*)(&As[buf][ar * LDT + ac]) = av;
        } else {
#if USE_ASYNC_LDS
            async_copy_b128<0>(Ah + k0 + ac, &As[buf][ar * LDT + ac]);
            async_copy_b128<16>(Ah + k0 + ac, &As[buf][ar * LDT + ac]);
#else
            v16h av = *(const v16h*)(Ah + k0 + ac);
            *(v16h*)(&As[buf][ar * LDT + ac]) = av;
#endif
        }
        // ---- B tile ----
#if USE_ASYNC_LDS
        if constexpr (BN == 128) {
            async_copy_b128<0>(Bp + k0 + bc, &Bs[buf][br * LDT + bc]);
            async_copy_b128<16>(Bp + k0 + bc, &Bs[buf][br * LDT + bc]);
        } else {
            async_copy_b128<0>(Bp + k0 + bc, &Bs[buf][br * LDT + bc]);
        }
#else
        if constexpr (BN == 128) {
            v16h bv = *(const v16h*)(Bp + k0 + bc);
            *(v16h*)(&Bs[buf][br * LDT + bc]) = bv;
        } else {
            v8h bv = *(const v8h*)(Bp + k0 + bc);
            *(v8h*)(&Bs[buf][br * LDT + bc]) = bv;
        }
#endif
    };

    // --- wave / lane geometry ------------------------------------------------
    const int wid  = tid >> 5;
    const int lane = tid & 31;
    const int wm   = (wid & 3) << 5;           // wave M offset (0..96)
    constexpr int WN  = BN / 2;
    constexpr int WNF = WN / 16;
    const int wn   = (wid >> 2) * WN;          // wave N offset
    const int lm   = lane & 15;                // M/N index within 16
    const int lh   = lane >> 4;                // lane half

    v8f acc[2][WNF];
    #pragma unroll
    for (int i = 0; i < 2; ++i)
        #pragma unroll
        for (int j = 0; j < WNF; ++j) acc[i][j] = v8f{};

    const int nk = K / BK;
    stage(0, 0);
    async_wait_all();
    __syncthreads();

    int buf = 0;
    for (int kt = 0; kt < nk; ++kt) {
        if (kt + 1 < nk) stage(kt + 1, buf ^ 1);

        // B fragments: 32x16 K-major per N row -> one contiguous 32B chunk
        v16h bfrag[WNF];
        #pragma unroll
        for (int j = 0; j < WNF; ++j) {
            const int nl = wn + j * 16 + lm;
            bfrag[j] = *(const v16h*)(&Bs[buf][nl * LDT + (lh << 4)]);
        }
        // A fragments: per lane two contiguous 8-half chunks (k=16*lh, 16*lh+8)
        #pragma unroll
        for (int i = 0; i < 2; ++i) {
            const int ml = wm + i * 16 + lm;
            const _Float16* ap = &As[buf][ml * LDT + (lh << 4)];
            v8h alo = *(const v8h*)(ap);
            v8h ahi = *(const v8h*)(ap + 8);
            v16h afrag = __builtin_shufflevector(
                alo, ahi, 0, 1, 2, 3, 4, 5, 6, 7, 8, 9, 10, 11, 12, 13, 14, 15);
            #pragma unroll
            for (int j = 0; j < WNF; ++j)
                acc[i][j] = __builtin_amdgcn_wmma_f32_16x16x32_f16(
                    false, afrag, false, bfrag[j],
                    (short)0, acc[i][j], false, false);
        }
        async_wait_all();
        __syncthreads();
        buf ^= 1;
    }

    // --- epilogue: alpha, bias, relu, fp32 and/or f16(layout) stores --------
    #pragma unroll
    for (int i = 0; i < 2; ++i) {
        #pragma unroll
        for (int j = 0; j < WNF; ++j) {
            const int n  = n0 + wn + j * 16 + lm;
            const float bv = bias ? bias[n] : 0.0f;
            #pragma unroll
            for (int r = 0; r < 8; ++r) {
                const int m = m0 + wm + i * 16 + (lh << 3) + r;
                float v = acc[i][j][r] * alpha + bv;
                if (relu) v = fmaxf(v, 0.0f);
                if (C32) C32[(long long)m * ldc32 + n] = v;
                if (Ch) {
                    long long off;
                    if (chmode == 0) {
                        off = (long long)m * ldch + n;
                    } else {
                        const int b = m >> 10, s = m & (kS - 1);
                        const int h = n >> 6,  dk = n & (kDK - 1);
                        if (chmode == 1)
                            off = ((((long long)b * kH + h) * kS) + s) * kDK + dk;
                        else
                            off = (((long long)b * kH + h) * kDK + dk) * (long long)kS + s;
                    }
                    Ch[off] = (_Float16)v;
                }
            }
        }
    }
}

// ---------------------------------------------------------------------------
// In-place row softmax over rows of length kS (one block per row)
// ---------------------------------------------------------------------------
__global__ __launch_bounds__(256) void softmax_rows(float* __restrict__ p)
{
    __shared__ float red[256];
    float* row = p + (long long)blockIdx.x * kS;
    const int t = threadIdx.x;

    float4 v = ((float4*)row)[t];
    float mx = fmaxf(fmaxf(v.x, v.y), fmaxf(v.z, v.w));
    red[t] = mx;
    __syncthreads();
    for (int s = 128; s > 0; s >>= 1) {
        if (t < s) red[t] = fmaxf(red[t], red[t + s]);
        __syncthreads();
    }
    mx = red[0];
    __syncthreads();

    v.x = expf(v.x - mx); v.y = expf(v.y - mx);
    v.z = expf(v.z - mx); v.w = expf(v.w - mx);
    red[t] = v.x + v.y + v.z + v.w;
    __syncthreads();
    for (int s = 128; s > 0; s >>= 1) {
        if (t < s) red[t] += red[t + s];
        __syncthreads();
    }
    const float inv = 1.0f / red[0];
    v.x *= inv; v.y *= inv; v.z *= inv; v.w *= inv;
    ((float4*)row)[t] = v;
}

// ---------------------------------------------------------------------------
// Fused residual add + LayerNorm over rows of length kD; optional f16 copy
// ---------------------------------------------------------------------------
__global__ __launch_bounds__(256) void layernorm_residual(
    const float* __restrict__ x, const float* __restrict__ r,
    const float* __restrict__ g, const float* __restrict__ b,
    float* __restrict__ out32, _Float16* __restrict__ outh)
{
    __shared__ float rsum[256];
    __shared__ float rsq[256];
    const long long base = (long long)blockIdx.x * kD;
    const int t = threadIdx.x;

    float4 xv = ((const float4*)(x + base))[t];
    float4 rv = ((const float4*)(r + base))[t];
    float e0 = xv.x + rv.x, e1 = xv.y + rv.y;
    float e2 = xv.z + rv.z, e3 = xv.w + rv.w;

    rsum[t] = e0 + e1 + e2 + e3;
    rsq[t]  = e0 * e0 + e1 * e1 + e2 * e2 + e3 * e3;
    __syncthreads();
    for (int s = 128; s > 0; s >>= 1) {
        if (t < s) { rsum[t] += rsum[t + s]; rsq[t] += rsq[t + s]; }
        __syncthreads();
    }
    const float mu  = rsum[0] * (1.0f / kD);
    const float var = rsq[0] * (1.0f / kD) - mu * mu;
    const float inv = rsqrtf(var + LN_EPS);

    float4 gv = ((const float4*)g)[t];
    float4 bv = ((const float4*)b)[t];
    float y0 = (e0 - mu) * inv * gv.x + bv.x;
    float y1 = (e1 - mu) * inv * gv.y + bv.y;
    float y2 = (e2 - mu) * inv * gv.z + bv.z;
    float y3 = (e3 - mu) * inv * gv.w + bv.w;

    ((float4*)(out32 + base))[t] = make_float4(y0, y1, y2, y3);
    if (outh) {
        const long long c = base + (long long)t * 4;
        outh[c + 0] = (_Float16)y0;
        outh[c + 1] = (_Float16)y1;
        outh[c + 2] = (_Float16)y2;
        outh[c + 3] = (_Float16)y3;
    }
}

// ---------------------------------------------------------------------------
// Weight preparation: out[C,R] f16 = transpose(in[R,C] f32); 32x32 LDS tiles
// ---------------------------------------------------------------------------
__global__ __launch_bounds__(256) void transpose_cvt(
    const float* __restrict__ in, _Float16* __restrict__ out, int R, int C)
{
    __shared__ float tile[32][33];
    const int bx = blockIdx.x << 5;   // col tile
    const int by = blockIdx.y << 5;   // row tile
    const int tx = threadIdx.x & 31;
    const int ty = threadIdx.x >> 5;  // 0..7

    #pragma unroll
    for (int i = 0; i < 4; ++i) {
        const int rr = by + ty + i * 8;
        tile[ty + i * 8][tx] = in[(long long)rr * C + bx + tx];
    }
    __syncthreads();
    #pragma unroll
    for (int i = 0; i < 4; ++i) {
        const int oc = bx + ty + i * 8;  // output row = original column
        out[(long long)oc * R + by + tx] = (_Float16)tile[tx][ty + i * 8];
    }
}

// ---------------------------------------------------------------------------
// Elementwise f32 -> f16 (n multiple of 1024)
// ---------------------------------------------------------------------------
__global__ __launch_bounds__(256) void cvt_f32_f16(
    const float* __restrict__ in, _Float16* __restrict__ out)
{
    const long long i = (long long)blockIdx.x * 1024 + (long long)threadIdx.x * 4;
    float4 v = *(const float4*)(in + i);
    out[i + 0] = (_Float16)v.x;
    out[i + 1] = (_Float16)v.y;
    out[i + 2] = (_Float16)v.z;
    out[i + 3] = (_Float16)v.w;
}

// ---------------------------------------------------------------------------
// Host-side orchestration
// ---------------------------------------------------------------------------
extern "C" void kernel_launch(void* const* d_in, const int* in_sizes, int n_in,
                              void* d_out, int out_size, void* d_ws, size_t ws_size,
                              hipStream_t stream)
{
    (void)in_sizes; (void)n_in; (void)out_size; (void)ws_size;

    const float* x   = (const float*)d_in[0];
    const float* Wq  = (const float*)d_in[1];
    const float* bq  = (const float*)d_in[2];
    const float* Wk  = (const float*)d_in[3];
    const float* bk  = (const float*)d_in[4];
    const float* Wv  = (const float*)d_in[5];
    const float* bv  = (const float*)d_in[6];
    const float* Wo  = (const float*)d_in[7];
    const float* bo  = (const float*)d_in[8];
    const float* W1  = (const float*)d_in[9];
    const float* b1  = (const float*)d_in[10];
    const float* W2  = (const float*)d_in[11];
    const float* b2  = (const float*)d_in[12];
    const float* g1  = (const float*)d_in[13];
    const float* be1 = (const float*)d_in[14];
    const float* g2  = (const float*)d_in[15];
    const float* be2 = (const float*)d_in[16];

    float* out_x = (float*)d_out;                                   // [B,S,D]
    float* attnW = (float*)d_out + (long long)kB * kS * kD;         // [B,H,S,S]

    // ---- carve workspace ----------------------------------------------------
    char* p = (char*)d_ws;
    auto carve = [&](size_t bytes) -> char* {
        char* r = p;
        p += (bytes + 255) & ~(size_t)255;
        return r;
    };
    const size_t NTOK = (size_t)kB * kS;       // 4096 token rows

    _Float16* WqT = (_Float16*)carve((size_t)kD * kD * 2);
    _Float16* WkT = (_Float16*)carve((size_t)kD * kD * 2);
    _Float16* WvT = (_Float16*)carve((size_t)kD * kD * 2);
    _Float16* WoT = (_Float16*)carve((size_t)kD * kD * 2);
    _Float16* W1T = (_Float16*)carve((size_t)kDFF * kD * 2);   // [DFF, D]
    _Float16* W2T = (_Float16*)carve((size_t)kD * kDFF * 2);   // [D, DFF]
    _Float16* xb  = (_Float16*)carve(NTOK * kD * 2);
    _Float16* qb  = (_Float16*)carve(NTOK * kD * 2);           // [B,H,S,DK]
    _Float16* kb  = (_Float16*)carve(NTOK * kD * 2);           // [B,H,S,DK]
    _Float16* vtb = (_Float16*)carve(NTOK * kD * 2);           // [B,H,DK,S]
    _Float16* ctxb = (_Float16*)carve(NTOK * kD * 2);          // [B,S,D] merged
    _Float16* x1b = (_Float16*)carve(NTOK * kD * 2);
    _Float16* hb  = (_Float16*)carve(NTOK * kDFF * 2);
    float* attn_out32 = (float*)carve(NTOK * kD * 4);
    float* x1         = (float*)carve(NTOK * kD * 4);
    float* ff32       = (float*)carve(NTOK * kD * 4);

    const dim3 blk(256);

    // ---- weight transpose + input conversion -------------------------------
    transpose_cvt<<<dim3(kD / 32, kD / 32), blk, 0, stream>>>(Wq, WqT, kD, kD);
    transpose_cvt<<<dim3(kD / 32, kD / 32), blk, 0, stream>>>(Wk, WkT, kD, kD);
    transpose_cvt<<<dim3(kD / 32, kD / 32), blk, 0, stream>>>(Wv, WvT, kD, kD);
    transpose_cvt<<<dim3(kD / 32, kD / 32), blk, 0, stream>>>(Wo, WoT, kD, kD);
    transpose_cvt<<<dim3(kDFF / 32, kD / 32), blk, 0, stream>>>(W1, W1T, kD, kDFF);
    transpose_cvt<<<dim3(kD / 32, kDFF / 32), blk, 0, stream>>>(W2, W2T, kDFF, kD);
    cvt_f32_f16<<<(unsigned)(NTOK * kD / 1024), blk, 0, stream>>>(x, xb);

    // ---- QKV projections (f16 head-layout outputs only) --------------------
    gemm_f16_wmma<128, false><<<dim3(kD / 128, NTOK / 128, 1), blk, 0, stream>>>(
        xb, WqT, bq, nullptr, 0, 0, qb, 0, 0, 0, 1, /*chmode=*/1,
        kD, 0, 0, 1.0f, 0);
    gemm_f16_wmma<128, false><<<dim3(kD / 128, NTOK / 128, 1), blk, 0, stream>>>(
        xb, WkT, bk, nullptr, 0, 0, kb, 0, 0, 0, 1, /*chmode=*/1,
        kD, 0, 0, 1.0f, 0);
    gemm_f16_wmma<128, false><<<dim3(kD / 128, NTOK / 128, 1), blk, 0, stream>>>(
        xb, WvT, bv, nullptr, 0, 0, vtb, 0, 0, 0, 1, /*chmode=*/2,
        kD, 0, 0, 1.0f, 0);

    // ---- scores = Q @ K^T / 8, written straight into d_out attn region -----
    gemm_f16_wmma<128, false><<<dim3(kS / 128, kS / 128, kB * kH), blk, 0, stream>>>(
        qb, kb, nullptr, attnW, kS, (long long)kS * kS,
        nullptr, 0, 0, 0, 1, 0,
        kDK, (long long)kS * kDK, (long long)kS * kDK, 0.125f, 0);

    // ---- softmax in place (d_out attn region == attn_weights output) -------
    softmax_rows<<<kB * kH * kS, blk, 0, stream>>>(attnW);

    // ---- ctx = P @ V  (A = fp32 probs, converted to f16 during staging) ----
    gemm_f16_wmma<64, true><<<dim3(1, kS / 128, kB * kH), blk, 0, stream>>>(
        attnW, vtb, nullptr, nullptr, 0, 0,
        ctxb, kD, (long long)kS * kD, (long long)kDK, kH, /*chmode=*/0,
        kS, (long long)kS * kS, (long long)kDK * kS, 1.0f, 0);

    // ---- attn_out = ctx @ Wo + bo ------------------------------------------
    gemm_f16_wmma<128, false><<<dim3(kD / 128, NTOK / 128, 1), blk, 0, stream>>>(
        ctxb, WoT, bo, attn_out32, kD, 0, nullptr, 0, 0, 0, 1, 0,
        kD, 0, 0, 1.0f, 0);

    // ---- x1 = LN(x + attn_out) ---------------------------------------------
    layernorm_residual<<<(unsigned)NTOK, blk, 0, stream>>>(
        x, attn_out32, g1, be1, x1, x1b);

    // ---- h = relu(x1 @ W1 + b1) --------------------------------------------
    gemm_f16_wmma<128, false><<<dim3(kDFF / 128, NTOK / 128, 1), blk, 0, stream>>>(
        x1b, W1T, b1, nullptr, 0, 0, hb, kDFF, 0, 0, 1, 0,
        kD, 0, 0, 1.0f, /*relu=*/1);

    // ---- ff = h @ W2 + b2 ---------------------------------------------------
    gemm_f16_wmma<128, false><<<dim3(kD / 128, NTOK / 128, 1), blk, 0, stream>>>(
        hb, W2T, b2, ff32, kD, 0, nullptr, 0, 0, 0, 1, 0,
        kDFF, 0, 0, 1.0f, 0);

    // ---- out = LN(x1 + ff) --------------------------------------------------
    layernorm_residual<<<(unsigned)NTOK, blk, 0, stream>>>(
        x1, ff32, g2, be2, out_x, nullptr);
}
